// Transformer_15547781611869
// MI455X (gfx1250) — matrix operand order repaired and checked by reading, not compiled, not run on hardware
//
#include <hip/hip_runtime.h>
#include <math.h>

// ---- problem constants ----
static constexpr int B   = 8;
static constexpr int L   = 2048;
static constexpr int D   = 128;
static constexpr int H   = 4;
static constexpr int DK  = 32;
static constexpr int DFF = 256;

typedef __attribute__((ext_vector_type(16))) __bf16 v16bf;
typedef __attribute__((ext_vector_type(8)))  float  v8f;
typedef __attribute__((ext_vector_type(4)))  unsigned uint4s;
typedef __attribute__((ext_vector_type(8)))  unsigned uint8s;

union V16 { v16bf v; uint4 q[2]; };

static __device__ __forceinline__ unsigned short f2bf_raw(float f) {
  union { float f; unsigned u; } a; a.f = f;
  unsigned u = a.u;
  u += 0x7FFFu + ((u >> 16) & 1u);          // round-to-nearest-even
  return (unsigned short)(u >> 16);
}

static __device__ __forceinline__ v8f wmma_bf16(v16bf a, v16bf b, v8f c) {
  // D = A(16x32 bf16) * B(32x16 bf16) + C(16x16 f32)
  return __builtin_amdgcn_wmma_f32_16x16x32_bf16(false, a, false, b, (short)0, c, false, false);
}

// A operand: 16x32 tile from row-major [16][ld] bf16 storage, K-offset k0.
static __device__ __forceinline__ v16bf load_A(const unsigned short* src, int ld, int k0, int lane) {
  int m = lane & 15, half = lane >> 4;
  const unsigned short* p = src + m * ld + k0 + half * 8;
  V16 r;
  r.q[0] = *(const uint4*)(p);
  r.q[1] = *(const uint4*)(p + 16);
  return r.v;
}

// B operand: 32x16 tile B[k][n] from TRANSPOSED storage srcT[n][k] (ld = ldk).
static __device__ __forceinline__ v16bf load_B_T(const unsigned short* srcT, int ldk, int n0, int k0, int lane) {
  int n = lane & 15, khalf = lane >> 4;
  const unsigned short* p = srcT + (n0 + n) * ldk + k0 + khalf * 8;
  V16 r;
  r.q[0] = *(const uint4*)(p);
  r.q[1] = *(const uint4*)(p + 16);
  return r.v;
}

// C layout: vgpr j -> row = j + 8*(lane/16), col = lane%16
static __device__ __forceinline__ void store_C_bf(v8f c, unsigned short* dst, int ld, int lane) {
  int half = lane >> 4, col = lane & 15;
  #pragma unroll
  for (int j = 0; j < 8; j++) dst[(j + 8 * half) * ld + col] = f2bf_raw(c[j]);
}

static __device__ __forceinline__ float rowmax16(float v) {
  #pragma unroll
  for (int m = 1; m < 16; m <<= 1) v = fmaxf(v, __shfl_xor(v, m, 32));
  return v;
}
static __device__ __forceinline__ float rowsum16(float v) {
  #pragma unroll
  for (int m = 1; m < 16; m <<= 1) v += __shfl_xor(v, m, 32);
  return v;
}

// ---- Tensor Data Mover: async DMA of a 32x32 bf16 tile (row stride rs elems) into LDS ----
// D# group0 (4 SGPRs) + group1 (8 SGPRs); 2D tile, data_size=2B, tensor dims == tile dims.
static __device__ __forceinline__ void tdm_load_tile32x32_bf16(unsigned long long gaddr,
                                                               unsigned lds_byte_off,
                                                               unsigned row_stride_elems) {
  uint4s g0;
  g0[0] = 1u;                                                  // count=1 (valid user D#)
  g0[1] = lds_byte_off;                                        // lds_addr
  g0[2] = (unsigned)gaddr;                                     // global_addr[31:0]
  g0[3] = ((unsigned)(gaddr >> 32) & 0x01FFFFFFu) | (2u << 30); // addr[56:32] | type=2
  uint8s g1;
  g1[0] = 0x00010000u;                                         // data_size=1 -> 2 bytes
  g1[1] = 32u << 16;                                           // tensor_dim0[15:0]=32
  g1[2] = 32u << 16;                                           // dim0_hi=0 | tensor_dim1[15:0]=32
  g1[3] = 32u << 16;                                           // dim1_hi=0 | tile_dim0=32
  g1[4] = 32u;                                                 // tile_dim1=32, tile_dim2=0
  g1[5] = row_stride_elems;                                    // tensor_dim0_stride[31:0]
  g1[6] = 0u;
  g1[7] = 0u;
  asm volatile("tensor_load_to_lds %0, %1" :: "s"(g0), "s"(g1) : "memory");
}

// ---------------- kernels ----------------

__global__ __launch_bounds__(256) void tk_init(const float* __restrict__ x,
                                               float* __restrict__ xf,
                                               unsigned short* __restrict__ xb, int n) {
  int i = blockIdx.x * 256 + threadIdx.x;
  if (i < n) { float v = x[i]; xf[i] = v; xb[i] = f2bf_raw(v); }
}

// QKV projection: x_bf16 [B*L,128] x W{q,k,v}[h] (128x32) -> q,k [B,H,L,32], v transposed [B,H,32,L]
__global__ __launch_bounds__(256) void tk_qkv(const unsigned short* __restrict__ xb,
                                              const float* __restrict__ Wq,
                                              const float* __restrict__ Wk,
                                              const float* __restrict__ Wv,
                                              unsigned short* __restrict__ qo,
                                              unsigned short* __restrict__ ko,
                                              unsigned short* __restrict__ vT) {
  extern __shared__ unsigned short sw[];           // [3][H][32][128] transposed weights (bf16)
  const int tid = threadIdx.x;
  const float* Ws[3] = {Wq, Wk, Wv};
  __builtin_prefetch(Wq + tid * 16, 0, 3);         // global_prefetch_b8 warm-up of weight panel
  for (int m = 0; m < 3; m++) {
    const float* W = Ws[m];
    unsigned short* dst = sw + m * (H * DK * D);
    for (int t = tid; t < H * DK * D; t += 256) {
      int h = t >> 12;
      int n = (t >> 7) & 31;
      int kk = t & 127;
      dst[t] = f2bf_raw(W[h * (D * DK) + kk * DK + n]);
    }
  }
  __syncthreads();

  const int b = blockIdx.x >> 4, rb = blockIdx.x & 15;
  const int w = tid >> 5, lane = tid & 31;
  const int r0 = rb * 128 + w * 16;

  v16bf A[4];
  const unsigned short* xrow = xb + ((size_t)b * L + r0) * D;
  #pragma unroll
  for (int ks = 0; ks < 4; ks++) A[ks] = load_A(xrow, D, ks * 32, lane);

  for (int h = 0; h < H; h++) {
    for (int m = 0; m < 3; m++) {
      const unsigned short* WT = sw + (m * H + h) * (DK * D);
      v8f c0 = {}; v8f c1 = {};
      #pragma unroll
      for (int ks = 0; ks < 4; ks++) {
        c0 = wmma_bf16(A[ks], load_B_T(WT, D, 0,  ks * 32, lane), c0);
        c1 = wmma_bf16(A[ks], load_B_T(WT, D, 16, ks * 32, lane), c1);
      }
      size_t base = ((size_t)(b * H + h) * L + r0) * DK;
      if (m == 0) {
        store_C_bf(c0, qo + base, DK, lane);
        store_C_bf(c1, qo + base + 16, DK, lane);
      } else if (m == 1) {
        store_C_bf(c0, ko + base, DK, lane);
        store_C_bf(c1, ko + base + 16, DK, lane);
      } else {
        int half = lane >> 4, col = lane & 15;
        unsigned short* vb = vT + (size_t)(b * H + h) * DK * L;
        #pragma unroll
        for (int j = 0; j < 8; j++) {
          int row = r0 + j + 8 * half;
          vb[(size_t)col * L + row]        = f2bf_raw(c0[j]);
          vb[(size_t)(col + 16) * L + row] = f2bf_raw(c1[j]);
        }
      }
    }
  }
}

// Flash attention per (b,h,128-query block). q/k [L,32], vT [32,L]. Output o_bf16 [B*L, H*DK].
// K/V tiles are streamed into per-wave double-buffered LDS slices by the Tensor Data Mover.
__global__ __launch_bounds__(256) void tk_attn(const unsigned short* __restrict__ qm,
                                               const unsigned short* __restrict__ km,
                                               const unsigned short* __restrict__ vTm,
                                               const float* __restrict__ mask,
                                               unsigned short* __restrict__ ob) {
  extern __shared__ __align__(16) unsigned short smem[];
  // per-wave: 2 bufs x (K tile 1024 us + V tile 1024 us) = 4096 us; pbuf after 8 waves.
  const int tid = threadIdx.x, w = tid >> 5, lane = tid & 31;
  const int half = lane >> 4, col = lane & 15;
  const int qb = blockIdx.x & 15;
  const int bh = blockIdx.x >> 4;          // = b*H + h
  const int h = bh & 3, b = bh >> 2;
  const int q0 = qb * 128 + w * 16;

  unsigned short* kvw = smem + w * 4096;                 // [buf][K|V][1024]
  unsigned short* pw  = smem + 8 * 4096 + w * 640;       // 16 x 40 padded P staging
  const unsigned ldsw = __builtin_amdgcn_groupstaticsize() + (unsigned)w * 8192u; // bytes

  const unsigned short* qptr = qm + ((size_t)bh * L + q0) * DK;
  const unsigned long long kga0 = (unsigned long long)(uintptr_t)(km  + (size_t)bh * L * DK);
  const unsigned long long vga0 = (unsigned long long)(uintptr_t)(vTm + (size_t)bh * DK * L);

  const v16bf Aq = load_A(qptr, DK, 0, lane);   // K = DK = 32: one WMMA per score tile

  float mq[8];
  #pragma unroll
  for (int j = 0; j < 8; j++) mq[j] = mask[(size_t)b * L + q0 + j + 8 * half];

  float mrow[8], lrow[8];
  #pragma unroll
  for (int j = 0; j < 8; j++) { mrow[j] = -INFINITY; lrow[j] = 0.0f; }
  v8f o0 = {}; v8f o1 = {};
  const float NEG = -3.402823466e38f;
  const int NKB = L / 32;

  // prologue: DMA tile 0 into buffer 0 (K tile: 32 keys x 64B rows; V tile: 32 d x 64B rows)
  tdm_load_tile32x32_bf16(kga0, ldsw, DK);
  tdm_load_tile32x32_bf16(vga0, ldsw + 2048u, L);

  for (int kb = 0; kb < NKB; kb++) {
    const int cur = kb & 1;
    const int key0 = kb * 32;
    if (kb + 1 < NKB) {
      // prefetch next tile into the other buffer, then wait for the current tile (2 older ops)
      unsigned nbuf = (unsigned)(1 - cur) * 4096u;
      tdm_load_tile32x32_bf16(kga0 + (unsigned long long)(kb + 1) * 2048ull, ldsw + nbuf, DK);
      tdm_load_tile32x32_bf16(vga0 + (unsigned long long)(kb + 1) * 64ull,  ldsw + nbuf + 2048u, L);
      __builtin_amdgcn_s_wait_tensorcnt(2);
    } else {
      __builtin_amdgcn_s_wait_tensorcnt(0);
    }
    const unsigned short* kslice = kvw + cur * 2048;       // [key][d] rows, ld=32
    const unsigned short* vslice = kslice + 1024;          // [d][key] rows, ld=32

    v16bf bk0 = load_B_T(kslice, DK, 0,  0, lane);
    v16bf bk1 = load_B_T(kslice, DK, 16, 0, lane);
    v8f z = {};
    v8f s0 = wmma_bf16(Aq, bk0, z);
    v8f s1 = wmma_bf16(Aq, bk1, z);

    float mk0 = mask[(size_t)b * L + key0 + col];
    float mk1 = mask[(size_t)b * L + key0 + 16 + col];

    #pragma unroll
    for (int j = 0; j < 8; j++) {
      float a0 = s0[j] + (1.0f - mq[j] * mk0) * NEG;
      float a1 = s1[j] + (1.0f - mq[j] * mk1) * NEG;
      float t = rowmax16(fmaxf(a0, a1));
      float mnew = fmaxf(mrow[j], t);
      float alpha = __expf(mrow[j] - mnew);
      float p0 = __expf(a0 - mnew);
      float p1 = __expf(a1 - mnew);
      lrow[j] = lrow[j] * alpha + rowsum16(p0 + p1);
      mrow[j] = mnew;
      o0[j] *= alpha; o1[j] *= alpha;
      int row = j + 8 * half;
      pw[row * 40 + col]      = f2bf_raw(p0);
      pw[row * 40 + col + 16] = f2bf_raw(p1);
    }
    // same-wave DS ordering + compiler barrier before re-reading P in A layout
    asm volatile("s_wait_dscnt 0" ::: "memory");
    v16bf Ap  = load_A(pw, 40, 0, lane);
    v16bf bv0 = load_B_T(vslice, DK, 0,  0, lane);
    v16bf bv1 = load_B_T(vslice, DK, 16, 0, lane);
    o0 = wmma_bf16(Ap, bv0, o0);
    o1 = wmma_bf16(Ap, bv1, o1);
    // retire this iteration's LDS reads before next iteration's TDM overwrites this buffer (WAR)
    asm volatile("s_wait_dscnt 0" ::: "memory");
  }

  unsigned short* op = ob + (size_t)b * L * D + h * DK;
  #pragma unroll
  for (int j = 0; j < 8; j++) {
    float inv = 1.0f / lrow[j];
    int row = q0 + j + 8 * half;
    op[(size_t)row * D + col]      = f2bf_raw(o0[j] * inv);
    op[(size_t)row * D + col + 16] = f2bf_raw(o1[j] * inv);
  }
}

// Output projection + residual avg: x = (o @ Wo + x) / 2; keeps f32 (d_out) and bf16 copies in sync.
__global__ __launch_bounds__(256) void tk_oproj(const unsigned short* __restrict__ ob,
                                                const float* __restrict__ Wo,
                                                float* __restrict__ xf,
                                                unsigned short* __restrict__ xbo) {
  extern __shared__ unsigned short WT[];   // [128][128] Wo^T bf16
  const int tid = threadIdx.x;
  __builtin_prefetch(Wo + tid * 16, 0, 3);
  for (int t = tid; t < D * D; t += 256) {
    int n = t >> 7, kk = t & 127;
    WT[t] = f2bf_raw(Wo[kk * D + n]);
  }
  __syncthreads();
  const int b = blockIdx.x >> 4, rb = blockIdx.x & 15;
  const int w = tid >> 5, lane = tid & 31, half = lane >> 4, col = lane & 15;
  const int r0 = rb * 128 + w * 16;

  v16bf A[4];
  const unsigned short* arow = ob + ((size_t)b * L + r0) * D;
  #pragma unroll
  for (int ks = 0; ks < 4; ks++) A[ks] = load_A(arow, D, ks * 32, lane);

  for (int nt = 0; nt < 8; nt++) {
    v8f c = {};
    #pragma unroll
    for (int ks = 0; ks < 4; ks++)
      c = wmma_bf16(A[ks], load_B_T(WT, D, nt * 16, ks * 32, lane), c);
    #pragma unroll
    for (int j = 0; j < 8; j++) {
      size_t g = (size_t)b * L + r0 + j + 8 * half;
      int cc = nt * 16 + col;
      float r = (c[j] + xf[g * D + cc]) * 0.5f;
      xf[g * D + cc] = r;
      xbo[g * D + cc] = f2bf_raw(r);
    }
  }
}

// FFN1: f = gelu(x @ W1 + b1)  (exact erf GELU), f stored bf16 [B*L, 256]
__global__ __launch_bounds__(256) void tk_ffn1(const unsigned short* __restrict__ xb,
                                               const float* __restrict__ W1,
                                               const float* __restrict__ b1,
                                               unsigned short* __restrict__ fb) {
  extern __shared__ unsigned short WT[];   // [256][128] W1^T bf16
  const int tid = threadIdx.x;
  __builtin_prefetch(W1 + tid * 16, 0, 3);
  for (int t = tid; t < DFF * D; t += 256) {
    int n = t >> 7, kk = t & 127;
    WT[t] = f2bf_raw(W1[kk * DFF + n]);
  }
  __syncthreads();
  const int b = blockIdx.x >> 4, rb = blockIdx.x & 15;
  const int w = tid >> 5, lane = tid & 31, half = lane >> 4, col = lane & 15;
  const int r0 = rb * 128 + w * 16;

  v16bf A[4];
  const unsigned short* arow = xb + ((size_t)b * L + r0) * D;
  #pragma unroll
  for (int ks = 0; ks < 4; ks++) A[ks] = load_A(arow, D, ks * 32, lane);

  for (int nt = 0; nt < 16; nt++) {
    v8f c = {};
    #pragma unroll
    for (int ks = 0; ks < 4; ks++)
      c = wmma_bf16(A[ks], load_B_T(WT, D, nt * 16, ks * 32, lane), c);
    #pragma unroll
    for (int j = 0; j < 8; j++) {
      int cc = nt * 16 + col;
      float v = c[j] + b1[cc];
      float g = 0.5f * v * (1.0f + erff(v * 0.70710678118654752f));
      size_t gi = (size_t)b * L + r0 + j + 8 * half;
      fb[gi * DFF + cc] = f2bf_raw(g);
    }
  }
}

// FFN2 + residual avg: x = (f @ W2 + b2 + x) / 2
__global__ __launch_bounds__(256) void tk_ffn2(const unsigned short* __restrict__ fb,
                                               const float* __restrict__ W2,
                                               const float* __restrict__ b2,
                                               float* __restrict__ xf,
                                               unsigned short* __restrict__ xbo) {
  extern __shared__ unsigned short WT[];   // [128][256] W2^T bf16
  const int tid = threadIdx.x;
  __builtin_prefetch(W2 + tid * 16, 0, 3);
  for (int t = tid; t < D * DFF; t += 256) {
    int n = t >> 8, kk = t & 255;
    WT[t] = f2bf_raw(W2[kk * D + n]);
  }
  __syncthreads();
  const int b = blockIdx.x >> 4, rb = blockIdx.x & 15;
  const int w = tid >> 5, lane = tid & 31, half = lane >> 4, col = lane & 15;
  const int r0 = rb * 128 + w * 16;

  v16bf A[8];
  const unsigned short* arow = fb + ((size_t)b * L + r0) * DFF;
  #pragma unroll
  for (int ks = 0; ks < 8; ks++) A[ks] = load_A(arow, DFF, ks * 32, lane);

  for (int nt = 0; nt < 8; nt++) {
    v8f c = {};
    #pragma unroll
    for (int ks = 0; ks < 8; ks++)
      c = wmma_bf16(A[ks], load_B_T(WT, DFF, nt * 16, ks * 32, lane), c);
    #pragma unroll
    for (int j = 0; j < 8; j++) {
      size_t g = (size_t)b * L + r0 + j + 8 * half;
      int cc = nt * 16 + col;
      float r = (c[j] + b2[cc] + xf[g * D + cc]) * 0.5f;
      xf[g * D + cc] = r;
      xbo[g * D + cc] = f2bf_raw(r);
    }
  }
}

// ---------------- host ----------------

extern "C" void kernel_launch(void* const* d_in, const int* in_sizes, int n_in,
                              void* d_out, int out_size, void* d_ws, size_t ws_size,
                              hipStream_t stream) {
  (void)in_sizes; (void)n_in; (void)out_size; (void)ws_size;
  const float* x    = (const float*)d_in[0];
  const float* mask = (const float*)d_in[1];
  const float* Wq   = (const float*)d_in[2];
  const float* Wk   = (const float*)d_in[3];
  const float* Wv   = (const float*)d_in[4];
  const float* Wo   = (const float*)d_in[5];
  const float* W1   = (const float*)d_in[6];
  const float* b1   = (const float*)d_in[7];
  const float* W2   = (const float*)d_in[8];
  const float* b2   = (const float*)d_in[9];
  float* xf = (float*)d_out;

  const size_t NTOK = (size_t)B * L;          // 16384
  unsigned short* xb = (unsigned short*)d_ws; // bf16 x          [NTOK*128]
  unsigned short* qb = xb + NTOK * D;         // q   [B,H,L,32]
  unsigned short* kb = qb + NTOK * D;         // k   [B,H,L,32]
  unsigned short* vT = kb + NTOK * D;         // v^T [B,H,32,L]
  unsigned short* ob = vT + NTOK * D;         // attn out bf16   [NTOK*128]
  unsigned short* fb = ob + NTOK * D;         // ffn mid bf16    [NTOK*256]

  const int lds_qkv  = 3 * H * DK * D * 2;            // 96 KB
  const int lds_attn = (8 * 4096 + 8 * 640) * 2;      // 74 KB (K/V double buffers + P staging)
  const int lds_oprj = D * D * 2;                     // 32 KB
  const int lds_ffn  = D * DFF * 2;                   // 64 KB
  hipFuncSetAttribute((const void*)tk_qkv,   hipFuncAttributeMaxDynamicSharedMemorySize, lds_qkv);
  hipFuncSetAttribute((const void*)tk_attn,  hipFuncAttributeMaxDynamicSharedMemorySize, lds_attn);
  hipFuncSetAttribute((const void*)tk_oproj, hipFuncAttributeMaxDynamicSharedMemorySize, lds_oprj);
  hipFuncSetAttribute((const void*)tk_ffn1,  hipFuncAttributeMaxDynamicSharedMemorySize, lds_ffn);
  hipFuncSetAttribute((const void*)tk_ffn2,  hipFuncAttributeMaxDynamicSharedMemorySize, lds_ffn);

  const int n = (int)(NTOK * D);
  tk_init<<<(n + 255) / 256, 256, 0, stream>>>(x, xf, xb, n);

  for (int i = 0; i < 2; i++) {
    tk_qkv<<<B * (L / 128), 256, lds_qkv, stream>>>(xb,
        Wq + (size_t)i * H * D * DK, Wk + (size_t)i * H * D * DK, Wv + (size_t)i * H * D * DK,
        qb, kb, vT);
    tk_attn<<<B * H * (L / 128), 256, lds_attn, stream>>>(qb, kb, vT, mask, ob);
    tk_oproj<<<B * (L / 128), 256, lds_oprj, stream>>>(ob, Wo + (size_t)i * D * D, xf, xb);
    tk_ffn1<<<B * (L / 128), 256, lds_ffn, stream>>>(xb, W1 + (size_t)i * D * DFF, b1 + (size_t)i * DFF, fb);
    tk_ffn2<<<B * (L / 128), 256, lds_ffn, stream>>>(fb, W2 + (size_t)i * D * DFF, b2 + (size_t)i * D, xf, xb);
  }
}
